// Quantizer_20736102105450
// MI455X (gfx1250) — matrix operand order repaired and checked by reading, not compiled, not run on hardware
//
#include <hip/hip_runtime.h>

typedef __attribute__((ext_vector_type(16))) __bf16        v16bf;
typedef __attribute__((ext_vector_type(8)))  float         v8f;
typedef __attribute__((ext_vector_type(4)))  float         f4;
typedef __attribute__((ext_vector_type(4)))  unsigned int  u4;
typedef __attribute__((ext_vector_type(2)))  unsigned int  u2;
typedef __attribute__((ext_vector_type(8)))  int           i8v;
typedef __attribute__((ext_vector_type(4)))  int           i4v;
typedef unsigned short u16;

#define DIM      256
#define KCODES   1024
#define HWSZ     4096
#define CHW      (256 * 4096)
#define SP       264          // padded LDS row stride (u16): 528B -> 4-bank step per row
#define CROWS    128          // codebook rows per LDS chunk
#define ZROWS    256          // z rows per workgroup (2 tiles of 128 -> 2x A-reuse)
#define NTHREADS 256

__device__ __forceinline__ u16 f2bf(float f) {
    unsigned u = __float_as_uint(f);
    u += 0x7fffu + ((u >> 16) & 1u);           // round-to-nearest-even
    return (u16)(u >> 16);
}

union Frag { v16bf v; u4 q[2]; };

// ---- TDM: 2D tile (256 x 128 rows, 2B elems) global -> LDS, padded 16B per 512B row
// D# built per CDNA5 ISA (groups 0/1; groups 2/3 zero = 2D).
__device__ __forceinline__ void tdm_load_chunk(const u16* gsrc, unsigned lds_off) {
    unsigned long long ga = (unsigned long long)gsrc;
    u4 g0;
    g0.x = 1u;                                            // count=1, user descriptor
    g0.y = lds_off;                                       // lds_addr (byte offset)
    g0.z = (unsigned)(ga & 0xffffffffu);                  // global_addr[31:0]
    g0.w = (unsigned)((ga >> 32) & 0x01ffffffu)           // global_addr[56:32]
         | (2u << 30);                                    // type = 2 ("image")
    i8v g1;
    g1[0] = (int)((1u << 16)                              // data_size: 2 bytes
                | (1u << 20)                              // pad_enable
                | (6u << 22)                              // pad_interval: 128 DWORDs (512B)
                | (3u << 25));                            // pad_amount: 4 DWORDs (16B)
    g1[1] = (int)((DIM & 0xffffu) << 16);                 // tensor_dim0 lo16 (=256)
    g1[2] = (int)((DIM >> 16) | ((KCODES & 0xffffu) << 16)); // dim0 hi | tensor_dim1 lo
    g1[3] = (int)((KCODES >> 16) | (DIM << 16));          // dim1 hi | tile_dim0 = 256
    g1[4] = (int)(CROWS);                                 // tile_dim1 = 128, tile_dim2 = 0
    g1[5] = (int)(DIM);                                   // tensor_dim0_stride lo32 = 256
    g1[6] = 0;                                            // stride0 hi | stride1 lo (unused 2D)
    g1[7] = 0;
    i4v gz = {0, 0, 0, 0};
#if defined(__clang_major__) && (__clang_major__ >= 23)
    i8v gz8 = {0, 0, 0, 0, 0, 0, 0, 0};
    __builtin_amdgcn_tensor_load_to_lds(g0, g1, gz, gz, gz8, 0);
#else
    __builtin_amdgcn_tensor_load_to_lds(g0, g1, gz, gz, 0);
#endif
}

// ---------------- Kernel 1: codebook fp32 -> bf16, plus half-squared-norms ----
__global__ void vq_prep(const float* __restrict__ cb,
                        u16* __restrict__ cbb, float* __restrict__ h) {
    int k = blockIdx.x, t = threadIdx.x;       // 1024 blocks x 256 threads
    float v = cb[k * DIM + t];
    cbb[k * DIM + t] = f2bf(v);
    float s = v * v;
    #pragma unroll
    for (int o = 16; o > 0; o >>= 1) s += __shfl_down(s, o, 32);
    __shared__ float red[8];
    if ((t & 31) == 0) red[t >> 5] = s;
    __syncthreads();
    if (t == 0) {
        float tot = 0.f;
        #pragma unroll
        for (int i = 0; i < 8; ++i) tot += red[i];
        h[k] = 0.5f * tot;
    }
}

// ---------------- Kernel 2: bf16 WMMA distance + argmin + gather -------------
__global__ void __launch_bounds__(NTHREADS)
vq_main(const float* __restrict__ z, const float* __restrict__ cbf,
        const u16* __restrict__ cbb, const float* __restrict__ hws,
        float* __restrict__ out) {
    // z-stage region; dead after B-frag hoist -> reused as codebook double buffer
    __shared__ __align__(16) u16 smem[2][CROWS * SP];
    __shared__ float hl[KCODES];
    __shared__ int   idxl[ZROWS];

    const int tid  = threadIdx.x;
    const int wv   = tid >> 5;
    const int lane = tid & 31;
    const int half = lane >> 4;
    const int lm   = lane & 15;
    const long nbase = (long)blockIdx.x * ZROWS;
    const int  bidx  = (int)(nbase / HWSZ);
    const int  p0    = (int)(nbase % HWSZ);

    for (int i = tid; i < KCODES; i += NTHREADS) hl[i] = hws[i];

    // ---- stage 256 z rows: z_flat[n][d] = z[b*CHW + d*HW + p], bf16 transposed ----
    const float* zb = z + (long)bidx * CHW + p0;
    for (int i = tid; i < 4096; i += NTHREADS) {
        int d0 = (i >> 6) << 2, r0 = (i & 63) << 2;
        f4 f0 = *(const f4*)(zb + (long)(d0 + 0) * HWSZ + r0);
        f4 f1 = *(const f4*)(zb + (long)(d0 + 1) * HWSZ + r0);
        f4 f2 = *(const f4*)(zb + (long)(d0 + 2) * HWSZ + r0);
        f4 f3 = *(const f4*)(zb + (long)(d0 + 3) * HWSZ + r0);
        u16* dst = &smem[r0 >> 7][0];
        int  rl  = r0 & 127;
        #pragma unroll
        for (int j = 0; j < 4; ++j) {
            unsigned lo = (unsigned)f2bf(f0[j]) | ((unsigned)f2bf(f1[j]) << 16);
            unsigned hi = (unsigned)f2bf(f2[j]) | ((unsigned)f2bf(f3[j]) << 16);
            *(u2*)&dst[(rl + j) * SP + d0] = (u2){lo, hi};
        }
    }
    __syncthreads();

    // ---- hoist 2 sets of B fragments (z as B-matrix 32x16), invariant over k ----
    // B layout: lane holds col n=lm; k = half*8+0..7 and 16+half*8+0..7 along d.
    Frag bf0[8], bf1[8];
    {
        const u16* zr0 = &smem[0][(wv * 16 + lm) * SP];
        const u16* zr1 = &smem[1][(wv * 16 + lm) * SP];
        #pragma unroll
        for (int dt = 0; dt < 8; ++dt) {
            bf0[dt].q[0] = *(const u4*)&zr0[dt * 32 + half * 8];
            bf0[dt].q[1] = *(const u4*)&zr0[dt * 32 + 16 + half * 8];
            bf1[dt].q[0] = *(const u4*)&zr1[dt * 32 + half * 8];
            bf1[dt].q[1] = *(const u4*)&zr1[dt * 32 + 16 + half * 8];
        }
    }
    asm volatile("s_wait_dscnt 0x0" ::: "memory");   // frags landed before TDM overwrite
    __syncthreads();

    const unsigned lds0 = (unsigned)(unsigned long long)(void*)&smem[0][0];
    const unsigned lds1 = (unsigned)(unsigned long long)(void*)&smem[1][0];
    if (wv == 0) {                                   // prime double buffer (TENSORcnt=2)
        tdm_load_chunk(cbb + 0 * CROWS * DIM, lds0);
        tdm_load_chunk(cbb + 1 * CROWS * DIM, lds1);
    }

    float minv0 = 3.4028235e38f, minv1 = 3.4028235e38f;
    int   mini0 = 0,             mini1 = 0;

    for (int kc = 0; kc < KCODES / CROWS; ++kc) {    // 8 chunks
        if (wv == 0) {
            if (kc < 7) __builtin_amdgcn_s_wait_tensorcnt(1);
            else        __builtin_amdgcn_s_wait_tensorcnt(0);
        }
        __syncthreads();

        const u16* buf = &smem[kc & 1][0];
        #pragma unroll 1
        for (int kt = 0; kt < 8; ++kt) {
            v8f acc0 = {0.f,0.f,0.f,0.f,0.f,0.f,0.f,0.f};
            v8f acc1 = {0.f,0.f,0.f,0.f,0.f,0.f,0.f,0.f};
            // A layout: lane holds row m=lm; k = half*8+0..7 and 16+half*8+0..7.
            const u16* ar = &buf[(kt * 16 + lm) * SP];
            #pragma unroll
            for (int dt = 0; dt < 8; ++dt) {
                Frag af;
                af.q[0] = *(const u4*)&ar[dt * 32 + half * 8];
                af.q[1] = *(const u4*)&ar[dt * 32 + 16 + half * 8];
                acc0 = __builtin_amdgcn_wmma_f32_16x16x32_bf16(
                           false, af.v, false, bf0[dt].v, (short)0, acc0, false, false);
                acc1 = __builtin_amdgcn_wmma_f32_16x16x32_bf16(
                           false, af.v, false, bf1[dt].v, (short)0, acc1, false, false);
            }
            const int kb = kc * CROWS + kt * 16 + half * 8;
            f4 h0 = *(const f4*)&hl[kb];
            f4 h1 = *(const f4*)&hl[kb + 4];
            float hs[8] = {h0.x, h0.y, h0.z, h0.w, h1.x, h1.y, h1.z, h1.w};
            #pragma unroll
            for (int j = 0; j < 8; ++j) {            // ascending k + strict '<' => first-min
                float s0 = hs[j] - acc0[j];
                if (s0 < minv0) { minv0 = s0; mini0 = kb + j; }
                float s1 = hs[j] - acc1[j];
                if (s1 < minv1) { minv1 = s1; mini1 = kb + j; }
            }
        }
        asm volatile("s_wait_dscnt 0x0" ::: "memory");
        __syncthreads();                             // chunk fully consumed
        if (wv == 0 && kc + 2 < KCODES / CROWS)
            tdm_load_chunk(cbb + (long)(kc + 2) * CROWS * DIM, (kc & 1) ? lds1 : lds0);
    }

    // ---- merge the two lane-halves holding the same z vector ----
    {
        float ov = __shfl_xor(minv0, 16, 32);
        int   oi = __shfl_xor(mini0, 16, 32);
        if (ov < minv0 || (ov == minv0 && oi < mini0)) { minv0 = ov; mini0 = oi; }
        ov = __shfl_xor(minv1, 16, 32);
        oi = __shfl_xor(mini1, 16, 32);
        if (ov < minv1 || (ov == minv1 && oi < mini1)) { minv1 = ov; mini1 = oi; }
        if (half == 0) {
            idxl[wv * 16 + lm]       = mini0;
            idxl[128 + wv * 16 + lm] = mini1;
        }
    }
    __syncthreads();

    // ---- gather fp32 code rows: out[n*256 + d] = codebook[idx[n]][d] ----
    const f4* cb4 = (const f4*)cbf;
    f4* o4 = (f4*)out + nbase * 64;
    for (int i = tid; i < ZROWS * 64; i += NTHREADS) {
        int r = i >> 6, c = i & 63;
        o4[r * 64 + c] = cb4[(long)idxl[r] * 64 + c];
    }
}

extern "C" void kernel_launch(void* const* d_in, const int* in_sizes, int n_in,
                              void* d_out, int out_size, void* d_ws, size_t ws_size,
                              hipStream_t stream) {
    const float* z  = (const float*)d_in[0];
    const float* cb = (const float*)d_in[1];
    u16*   cbb = (u16*)d_ws;                                   // 512 KB bf16 codebook
    float* h   = (float*)((char*)d_ws + (size_t)KCODES * DIM * sizeof(u16)); // 4 KB
    float* out = (float*)d_out;

    vq_prep<<<KCODES, NTHREADS, 0, stream>>>(cb, cbb, h);
    vq_main<<<(32 * 64 * 64) / ZROWS, NTHREADS, 0, stream>>>(z, cb, cbb, h, out);
}
